// MMPENetwork_21423296872540
// MI455X (gfx1250) — compile-verified, matrix operand order checked by reading, not run on hardware
//
#include <hip/hip_runtime.h>
#include <hip/hip_bf16.h>
#include <math.h>

typedef __attribute__((ext_vector_type(16))) _Float16 v16h;
typedef __attribute__((ext_vector_type(8)))  _Float16 v8h;
typedef __attribute__((ext_vector_type(8)))  float    v8f;

#define NPIX  5456      // total concat pixels per (b, channel)
#define BSZ   16
#define CCH   256
#define KDIM  2304      // 9 taps * 256 ci
#define OUTCH 69
#define OCPAD 80        // 69 padded to 5 tiles of 16

// ---------------------------------------------------------------------------
// Utility: zero fill (for activation halos)
// ---------------------------------------------------------------------------
__global__ __launch_bounds__(256)
void k_zero(uint4* __restrict__ p, size_t n16)
{
    size_t t = (size_t)blockIdx.x * 256 + threadIdx.x;
    if (t < n16) p[t] = make_uint4(0u, 0u, 0u, 0u);
}

// ---------------------------------------------------------------------------
// Layout converters
// ---------------------------------------------------------------------------

// NCHW f32 -> padded NHWC f16 ([b][H+2][W+2][256], interior only; halo stays 0)
__global__ __launch_bounds__(256)
void k_cvt_act(const float* __restrict__ src, _Float16* __restrict__ dst,
               int H, int W, int total)
{
    int t = blockIdx.x * 256 + threadIdx.x;
    if (t >= total) return;
    int x = t % W; int r = t / W;
    int y = r % H; r /= H;
    int c = r & 255; int b = r >> 8;
    const int Hp = H + 2, Wp = W + 2;
    dst[(((size_t)(b * Hp + y + 1) * Wp + (x + 1)) << 8) + c] = (_Float16)src[t];
}

// dw (7,256oc,256ci,3,3) f32 -> w16[l][oc][tap*256+ci] f16
__global__ __launch_bounds__(256)
void k_cvt_w(const float* __restrict__ dw, _Float16* __restrict__ w16, int total)
{
    int t = blockIdx.x * 256 + threadIdx.x;
    if (t >= total) return;
    int k = t % KDIM; int r = t / KDIM;
    int oc = r & 255; int l = r >> 8;
    int tap = k >> 8; int ci = k & 255;
    int ky = tap / 3, kx = tap - ky * 3;
    w16[t] = (_Float16)dw[((((size_t)l * CCH + oc) * CCH + ci) * 3 + ky) * 3 + kx];
}

// wf (69,256,3,3) f32 -> wf16[80][2304] f16, zero-padded oc 69..79
__global__ __launch_bounds__(256)
void k_cvt_wf(const float* __restrict__ wf, _Float16* __restrict__ w16, int total)
{
    int t = blockIdx.x * 256 + threadIdx.x;
    if (t >= total) return;
    int k = t % KDIM; int oc = t / KDIM;
    float v = 0.f;
    if (oc < OUTCH) {
        int tap = k >> 8; int ci = k & 255;
        int ky = tap / 3, kx = tap - ky * 3;
        v = wf[(((size_t)oc * CCH + ci) * 3 + ky) * 3 + kx];
    }
    w16[t] = (_Float16)v;
}

// ---------------------------------------------------------------------------
// Main conv layer: implicit GEMM over halo-padded input, branchless K loop.
// Per wave: 64 oc x 32 pixels (4 A-frags x 2 B-frags -> 8 WMMA per K-chunk).
// Workgroup (8 waves): 256 oc x 64 pixels.
// ---------------------------------------------------------------------------
template<int H, int W>
__global__ __launch_bounds__(256)
void k_conv_silu(const _Float16* __restrict__ in,   // [B][H+2][W+2][256]
                 const _Float16* __restrict__ wgt,  // [256][2304]
                 const float*    __restrict__ bias, // [256]
                 _Float16*       __restrict__ out)  // [B][H+2][W+2][256]
{
    constexpr int Hp = H + 2, Wp = W + 2;
    constexpr int hw = H * W;

    const int lane = threadIdx.x & 31;
    const int wave = threadIdx.x >> 5;
    const int mblk = wave & 3;                    // oc block: mblk*64
    const int pgrp = (blockIdx.x << 1) + (wave >> 2);  // group of 32 pixels
    const int col  = lane & 15;
    const int hi   = lane >> 4;                   // K half-block select

    const int pix0 = pgrp * 32 + col;
    const int pix1 = pix0 + 16;
    const int b0 = pix0 / hw, r0 = pix0 - b0 * hw, y0 = r0 / W, x0 = r0 - y0 * W;
    const int b1 = pix1 / hw, r1 = pix1 - b1 * hw, y1 = r1 / W, x1 = r1 - y1 * W;

    // center-pixel pointers (interior of padded buffer)
    const _Float16* pc0 =
        in + (((size_t)(b0 * Hp + y0 + 1) * Wp + (x0 + 1)) << 8) + hi * 8;
    const _Float16* pc1 =
        in + (((size_t)(b1 * Hp + y1 + 1) * Wp + (x1 + 1)) << 8) + hi * 8;

    v8f zero = {};
    v8f acc[4][2];
#pragma unroll
    for (int m = 0; m < 4; ++m) { acc[m][0] = zero; acc[m][1] = zero; }

    const _Float16* wbase = wgt + (size_t)(mblk * 64 + col) * KDIM + hi * 8;

    for (int ky = 0; ky < 3; ++ky) {
        for (int kx = 0; kx < 3; ++kx) {
            const int doff = (((ky - 1) * Wp) + (kx - 1)) << 8;
            const _Float16* p0 = pc0 + doff;
            const _Float16* p1 = pc1 + doff;
            const _Float16* wt = wbase + ((ky * 3 + kx) << 8);
#pragma unroll
            for (int cc = 0; cc < 8; ++cc) {      // K chunk of 32 within tap
                const int ko = cc * 32;
                v16h bf0, bf1;
                *(v8h*)&bf0       = *(const v8h*)(p0 + ko);
                *((v8h*)&bf0 + 1) = *(const v8h*)(p0 + ko + 16);
                *(v8h*)&bf1       = *(const v8h*)(p1 + ko);
                *((v8h*)&bf1 + 1) = *(const v8h*)(p1 + ko + 16);
#pragma unroll
                for (int m = 0; m < 4; ++m) {
                    const _Float16* wr = wt + (size_t)m * 16 * KDIM + ko;
                    v16h af;
                    *(v8h*)&af       = *(const v8h*)(wr);
                    *((v8h*)&af + 1) = *(const v8h*)(wr + 16);
                    acc[m][0] = __builtin_amdgcn_wmma_f32_16x16x32_f16(
                        false, af, false, bf0, (short)0, acc[m][0], false, false);
                    acc[m][1] = __builtin_amdgcn_wmma_f32_16x16x32_f16(
                        false, af, false, bf1, (short)0, acc[m][1], false, false);
                }
            }
        }
    }

    // Epilogue: bias + SiLU, channels-last store (8 contiguous oc = 16B)
    const size_t ob0 = ((size_t)(b0 * Hp + y0 + 1) * Wp + (x0 + 1)) << 8;
    const size_t ob1 = ((size_t)(b1 * Hp + y1 + 1) * Wp + (x1 + 1)) << 8;
#pragma unroll
    for (int m = 0; m < 4; ++m) {
        const int oc0 = mblk * 64 + m * 16 + hi * 8;   // D: M = r + 8*hi
        v8h pk0, pk1;
#pragma unroll
        for (int r = 0; r < 8; ++r) {
            float bv = bias[oc0 + r];
            float v0 = acc[m][0][r] + bv;
            float v1 = acc[m][1][r] + bv;
            v0 = v0 / (1.f + __expf(-v0));
            v1 = v1 / (1.f + __expf(-v1));
            pk0[r] = (_Float16)v0;
            pk1[r] = (_Float16)v1;
        }
        *(v8h*)(out + ob0 + oc0) = pk0;
        *(v8h*)(out + ob1 + oc0) = pk1;
    }
}

// ---------------------------------------------------------------------------
// Final conv: 256 -> 69 (padded 80), writes f32 head[b][oc][NPIX].
// One wave per 16oc x 16pix tile, halo-padded input, branchless.
// ---------------------------------------------------------------------------
template<int H, int W>
__global__ __launch_bounds__(32)
void k_conv_final(const _Float16* __restrict__ in,   // [B][H+2][W+2][256]
                  const _Float16* __restrict__ wgt,  // [80][2304]
                  const float*    __restrict__ bias, // [69]
                  float*          __restrict__ head, // [B][69][NPIX]
                  int soff)
{
    constexpr int Hp = H + 2, Wp = W + 2;
    constexpr int hw = H * W;

    const int lane = threadIdx.x & 31;
    const int col  = lane & 15;
    const int hi   = lane >> 4;
    const int octile = blockIdx.y;
    const int pix  = blockIdx.x * 16 + col;
    const int b    = pix / hw;
    const int rem  = pix - b * hw;
    const int y    = rem / W;
    const int x    = rem - y * W;

    const _Float16* pc =
        in + (((size_t)(b * Hp + y + 1) * Wp + (x + 1)) << 8) + hi * 8;

    v8f acc = {};
    const _Float16* wbase = wgt + (size_t)(octile * 16 + col) * KDIM + hi * 8;

    for (int ky = 0; ky < 3; ++ky) {
        for (int kx = 0; kx < 3; ++kx) {
            const int doff = (((ky - 1) * Wp) + (kx - 1)) << 8;
            const _Float16* p = pc + doff;
            const _Float16* wt = wbase + ((ky * 3 + kx) << 8);
#pragma unroll
            for (int cc = 0; cc < 8; ++cc) {
                const int ko = cc * 32;
                v16h bf, af;
                *(v8h*)&bf       = *(const v8h*)(p + ko);
                *((v8h*)&bf + 1) = *(const v8h*)(p + ko + 16);
                *(v8h*)&af       = *(const v8h*)(wt + ko);
                *((v8h*)&af + 1) = *(const v8h*)(wt + ko + 16);
                acc = __builtin_amdgcn_wmma_f32_16x16x32_f16(
                    false, af, false, bf, (short)0, acc, false, false);
            }
        }
    }

#pragma unroll
    for (int r = 0; r < 8; ++r) {
        const int oc = octile * 16 + hi * 8 + r;
        if (oc < OUTCH)
            head[((size_t)(b * OUTCH + oc)) * NPIX + soff + rem] =
                acc[r] + bias[oc];
    }
}

// ---------------------------------------------------------------------------
// Post-processing
// ---------------------------------------------------------------------------

// fg = sigmoid(o1); pi = fg / sum_n(fg). One block per batch item.
__global__ __launch_bounds__(256)
void k_post_pi(const float* __restrict__ head,
               float* __restrict__ pi, float* __restrict__ fg)
{
    const int b = blockIdx.x, tid = threadIdx.x;
    __shared__ float red[256];
    const float* o1 = head + (size_t)b * OUTCH * NPIX;   // channel 0
    float loc[22];
    float s = 0.f;
#pragma unroll
    for (int i = 0; i < 22; ++i) {
        int n = tid + i * 256;
        float v = 0.f;
        if (n < NPIX) v = 1.f / (1.f + __expf(-o1[n]));
        loc[i] = v; s += v;
    }
    red[tid] = s; __syncthreads();
    for (int st = 128; st > 0; st >>= 1) {
        if (tid < st) red[tid] += red[tid + st];
        __syncthreads();
    }
    const float inv = 1.f / red[0];
#pragma unroll
    for (int i = 0; i < 22; ++i) {
        int n = tid + i * 256;
        if (n < NPIX) {
            fg[(size_t)b * NPIX + n] = loc[i];
            pi[(size_t)b * NPIX + n] = loc[i] * inv;
        }
    }
}

// mu = o2*scale + coord; bbox w/h over 17 joints; sig = max(softplus(o3), s*std)
__global__ __launch_bounds__(256)
void k_post_musig(const float* __restrict__ head,
                  const float* __restrict__ defc,   // (B,34,NPIX)
                  const float* __restrict__ osc,    // (B,34,NPIX)
                  const float* __restrict__ stdf,   // (1,34,1)
                  float* __restrict__ mu_o, float* __restrict__ sig_o)
{
    int t = blockIdx.x * 256 + threadIdx.x;
    if (t >= BSZ * NPIX) return;
    const int b = t / NPIX;
    const int n = t - b * NPIX;
    const float* hb = head + (size_t)b * OUTCH * NPIX;

    float mu[34];
    float mnx = 1e30f, mxx = -1e30f, mny = 1e30f, mxy = -1e30f;
#pragma unroll
    for (int j = 0; j < 34; ++j) {
        size_t ji = ((size_t)b * 34 + j) * NPIX + n;
        float m = hb[(size_t)(1 + j) * NPIX + n] * osc[ji] + defc[ji];
        mu[j] = m;
        if (j & 1) { mny = fminf(mny, m); mxy = fmaxf(mxy, m); }
        else       { mnx = fminf(mnx, m); mxx = fmaxf(mxx, m); }
    }
    const float sx = fmaxf(mxx - mnx, 1e-6f);
    const float sy = fmaxf(mxy - mny, 1e-6f);
#pragma unroll
    for (int j = 0; j < 34; ++j) {
        size_t ji = ((size_t)b * 34 + j) * NPIX + n;
        mu_o[ji] = mu[j];
        float o3 = hb[(size_t)(35 + j) * NPIX + n];
        float sp = (o3 > 20.f) ? o3 : log1pf(__expf(o3));
        sig_o[ji] = fmaxf(sp, ((j & 1) ? sy : sx) * stdf[j]);
    }
}

// ---------------------------------------------------------------------------
// Launch
// ---------------------------------------------------------------------------
static void launch_conv(int s, const _Float16* in, const _Float16* w,
                        const float* b, _Float16* out, hipStream_t st)
{
    switch (s) {
    case 0: k_conv_silu<64,64><<<BSZ*64*64/64, 256, 0, st>>>(in, w, b, out); break;
    case 1: k_conv_silu<32,32><<<BSZ*32*32/64, 256, 0, st>>>(in, w, b, out); break;
    case 2: k_conv_silu<16,16><<<BSZ*16*16/64, 256, 0, st>>>(in, w, b, out); break;
    case 3: k_conv_silu< 8, 8><<<BSZ* 8* 8/64, 256, 0, st>>>(in, w, b, out); break;
    default:k_conv_silu< 4, 4><<<BSZ* 4* 4/64, 256, 0, st>>>(in, w, b, out); break;
    }
}

static void launch_final(int s, const _Float16* in, const _Float16* w,
                         const float* b, float* head, int soff, hipStream_t st)
{
    switch (s) {
    case 0: k_conv_final<64,64><<<dim3(BSZ*64*64/16, OCPAD/16), 32, 0, st>>>(in, w, b, head, soff); break;
    case 1: k_conv_final<32,32><<<dim3(BSZ*32*32/16, OCPAD/16), 32, 0, st>>>(in, w, b, head, soff); break;
    case 2: k_conv_final<16,16><<<dim3(BSZ*16*16/16, OCPAD/16), 32, 0, st>>>(in, w, b, head, soff); break;
    case 3: k_conv_final< 8, 8><<<dim3(BSZ* 8* 8/16, OCPAD/16), 32, 0, st>>>(in, w, b, head, soff); break;
    default:k_conv_final< 4, 4><<<dim3(BSZ* 4* 4/16, OCPAD/16), 32, 0, st>>>(in, w, b, head, soff); break;
    }
}

extern "C" void kernel_launch(void* const* d_in, const int* in_sizes, int n_in,
                              void* d_out, int out_size, void* d_ws, size_t ws_size,
                              hipStream_t stream)
{
    (void)in_sizes; (void)n_in; (void)out_size; (void)ws_size;

    const float* f[5];
    for (int s = 0; s < 5; ++s) f[s] = (const float*)d_in[s];
    const float* dw   = (const float*)d_in[5];
    const float* db   = (const float*)d_in[6];
    const float* wf   = (const float*)d_in[7];
    const float* bf   = (const float*)d_in[8];
    const float* defc = (const float*)d_in[9];
    const float* osc  = (const float*)d_in[10];
    const float* stdf = (const float*)d_in[11];

    static const int Hs[5]    = {64, 32, 16, 8, 4};
    static const int pref[5]  = {0, 4096, 5120, 5376, 5440};       // logical concat
    static const int ppref[5] = {0, 4356, 5512, 5836, 5936};       // padded (H+2)(W+2)
    const int PPIX = 5972;                                          // padded total

    // Workspace carve-up
    char* ws = (char*)d_ws;
    const size_t szAct = (size_t)BSZ * PPIX * CCH * sizeof(_Float16); // 48.9 MB
    const size_t szW   = (size_t)7 * CCH * KDIM * sizeof(_Float16);   // 8.3 MB
    const size_t szWF  = (size_t)OCPAD * KDIM * sizeof(_Float16);
    _Float16* X    = (_Float16*)(ws);
    _Float16* Y    = (_Float16*)(ws + szAct);
    _Float16* W16  = (_Float16*)(ws + 2 * szAct);
    _Float16* WF16 = (_Float16*)(ws + 2 * szAct + szW);
    float*    HEAD = (float*)   (ws + 2 * szAct + szW + szWF);

    // Zero both padded activation buffers (halos must be 0); ~98MB, ~4us @ HBM peak
    {
        size_t n16 = (2 * szAct) / 16;
        k_zero<<<(unsigned)((n16 + 255) / 256), 256, 0, stream>>>((uint4*)ws, n16);
    }

    // Layout conversions
    for (int s = 0; s < 5; ++s) {
        int H = Hs[s], W = Hs[s];
        int tot = BSZ * CCH * H * W;
        k_cvt_act<<<(tot + 255) / 256, 256, 0, stream>>>(
            f[s], X + (size_t)BSZ * ppref[s] * CCH, H, W, tot);
    }
    {
        int tot = 7 * CCH * KDIM;
        k_cvt_w<<<(tot + 255) / 256, 256, 0, stream>>>(dw, W16, tot);
    }
    {
        int tot = OCPAD * KDIM;
        k_cvt_wf<<<(tot + 255) / 256, 256, 0, stream>>>(wf, WF16, tot);
    }

    // 7 fused conv+SiLU layers, ping-pong X<->Y (halos stay zero: interior-only writes)
    _Float16* cur = X;
    _Float16* nxt = Y;
    for (int l = 0; l < 7; ++l) {
        for (int s = 0; s < 5; ++s) {
            size_t aoff = (size_t)BSZ * ppref[s] * CCH;
            launch_conv(s, cur + aoff, W16 + (size_t)l * CCH * KDIM,
                        db + l * CCH, nxt + aoff, stream);
        }
        _Float16* t = cur; cur = nxt; nxt = t;
    }

    // Final head conv into f32 staging
    for (int s = 0; s < 5; ++s) {
        size_t aoff = (size_t)BSZ * ppref[s] * CCH;
        launch_final(s, cur + aoff, WF16, bf, HEAD, pref[s], stream);
    }

    // Outputs: pi | mu | sig | fg (flat, f32)
    float* out     = (float*)d_out;
    float* out_pi  = out;
    float* out_mu  = out_pi + (size_t)BSZ * NPIX;
    float* out_sig = out_mu + (size_t)BSZ * 34 * NPIX;
    float* out_fg  = out_sig + (size_t)BSZ * 34 * NPIX;

    k_post_pi<<<BSZ, 256, 0, stream>>>(HEAD, out_pi, out_fg);
    {
        int tot = BSZ * NPIX;
        k_post_musig<<<(tot + 255) / 256, 256, 0, stream>>>(
            HEAD, defc, osc, stdf, out_mu, out_sig);
    }
}
